// RG_SA_65807488909833
// MI455X (gfx1250) — matrix operand order repaired and checked by reading, not compiled, not run on hardware
//
#include <hip/hip_runtime.h>
#include <hip/hip_bf16.h>

typedef __attribute__((ext_vector_type(16))) __bf16 v16bf;
typedef __attribute__((ext_vector_type(8)))  float  v8f;

union FragU { uint4 u4[2]; v16bf v; };

static __device__ __forceinline__ unsigned short f2bf(float f) {
    unsigned u = __float_as_uint(f);
    u += 0x7FFFu + ((u >> 16) & 1u);       // round-to-nearest-even
    return (unsigned short)(u >> 16);
}

// ---------------------------------------------------------------- f32 -> bf16
__global__ void cvt_bf16_kernel(const float* __restrict__ src,
                                unsigned short* __restrict__ dst, long long n) {
    long long i = (long long)blockIdx.x * blockDim.x + threadIdx.x;
    long long stride = (long long)gridDim.x * blockDim.x;
    for (; i < n; i += stride) dst[i] = f2bf(src[i]);
}

// ------------------------------------------------- fused double stride-4 conv
// two depthwise 4x4 stride-4 convs (same weights) == one 16x16 patch reduction
__global__ __launch_bounds__(192)
void reduce_kernel(const float* __restrict__ x, const float* __restrict__ red_w,
                   const float* __restrict__ red_b, float* __restrict__ s2) {
    int c  = threadIdx.x;              // channel 0..191
    int h2 = blockIdx.x, w2 = blockIdx.y, b = blockIdx.z;
    float rw[16];
    float sw = 0.f;
#pragma unroll
    for (int j = 0; j < 16; ++j) { rw[j] = red_w[c * 16 + j]; sw += rw[j]; }
    float acc = 0.f;
    const float* xb = x + (size_t)b * 16384 * 192 + c;
    for (int dy = 0; dy < 16; ++dy) {
        int yy = h2 * 16 + dy;
        for (int dx = 0; dx < 16; ++dx) {
            int xx = w2 * 16 + dx;
            float v = xb[(size_t)(yy * 128 + xx) * 192];  // coalesced across c
            acc += v * rw[(dy >> 2) * 4 + (dx >> 2)] * rw[(dy & 3) * 4 + (dx & 3)];
        }
    }
    float bs  = red_b[c];
    s2[((size_t)b * 192 + c) * 64 + h2 * 8 + w2] = acc + bs * sw + bs;
}

// ---------------------------- dw3x3 + 1x1 conv + LayerNorm + exact GELU per b
__global__ __launch_bounds__(256)
void mix_kernel(const float* __restrict__ s2, const float* __restrict__ dw_w,
                const float* __restrict__ dw_b, const float* __restrict__ conv_w,
                const float* __restrict__ conv_b, const float* __restrict__ ln_g,
                const float* __restrict__ ln_b, float* __restrict__ xg) {
    __shared__ float s_img[192 * 64];
    __shared__ float t_img[192 * 64];
    __shared__ float xo[64 * 193];
    int b = blockIdx.x, tid = threadIdx.x;
    for (int i = tid; i < 192 * 64; i += 256) s_img[i] = s2[(size_t)b * 192 * 64 + i];
    __syncthreads();
    for (int i = tid; i < 192 * 64; i += 256) {
        int c = i / 64, pos = i % 64, y = pos >> 3, xp = pos & 7;
        float acc = dw_b[c];
        for (int ky = 0; ky < 3; ++ky) {
            int yy = y + ky - 1; if ((unsigned)yy >= 8u) continue;
            for (int kx = 0; kx < 3; ++kx) {
                int xx = xp + kx - 1; if ((unsigned)xx >= 8u) continue;
                acc += s_img[c * 64 + yy * 8 + xx] * dw_w[c * 9 + ky * 3 + kx];
            }
        }
        t_img[i] = acc;
    }
    __syncthreads();
    for (int i = tid; i < 64 * 192; i += 256) {
        int pos = i / 192, o = i % 192;
        float acc = conv_b[o];
        for (int c = 0; c < 192; ++c) acc += conv_w[o * 192 + c] * t_img[c * 64 + pos];
        xo[pos * 193 + o] = acc;
    }
    __syncthreads();
    if (tid < 64) {
        int pos = tid;
        float mu = 0.f;
        for (int c = 0; c < 192; ++c) mu += xo[pos * 193 + c];
        mu *= (1.f / 192.f);
        float var = 0.f;
        for (int c = 0; c < 192; ++c) { float d = xo[pos * 193 + c] - mu; var += d * d; }
        var *= (1.f / 192.f);
        float inv = rsqrtf(var + 1e-5f);
        for (int c = 0; c < 192; ++c) {
            float v = (xo[pos * 193 + c] - mu) * inv * ln_g[c] + ln_b[c];
            float g = 0.5f * v * (1.f + erff(v * 0.70710678118654752f));  // exact GELU
            xg[((size_t)b * 64 + pos) * 192 + c] = g;
        }
    }
}

// -------------------------------------- q/k projections + channel-attn softmax
__global__ __launch_bounds__(256)
void qk_attn_kernel(const float* __restrict__ xg, const float* __restrict__ q_w,
                    const float* __restrict__ k_w, float* __restrict__ attn) {
    __shared__ float xs[64 * 192];
    __shared__ float qs[192 * 64];   // [o][m]
    __shared__ float ks[96 * 64];    // [r][m]
    __shared__ float ls[2304];       // [(h*24+d)][e]
    int b = blockIdx.x, tid = threadIdx.x;
    for (int i = tid; i < 64 * 192; i += 256) xs[i] = xg[(size_t)b * 64 * 192 + i];
    __syncthreads();
    for (int i = tid; i < 192 * 64; i += 256) {
        int o = i / 64, m = i % 64;
        float acc = 0.f;
        for (int c = 0; c < 192; ++c) acc += xs[m * 192 + c] * q_w[o * 192 + c];
        qs[o * 64 + m] = acc;
    }
    for (int i = tid; i < 96 * 64; i += 256) {
        int r = i / 64, m = i % 64;
        float acc = 0.f;
        for (int c = 0; c < 192; ++c) acc += xs[m * 192 + c] * k_w[r * 192 + c];
        ks[r * 64 + m] = acc;
    }
    __syncthreads();
    for (int idx = tid; idx < 2304; idx += 256) {
        int h = idx / 288, rem = idx % 288, d = rem / 12, e = rem % 12;
        const float* qp = &qs[(h * 24 + d) * 64];
        const float* kp = &ks[(h * 12 + e) * 64];
        float acc = 0.f;
        for (int m = 0; m < 64; ++m) acc += qp[m] * kp[m];
        ls[idx] = acc * 0.28867513459481287f;   // (HD*0.5)^-0.5 = 12^-0.5
    }
    __syncthreads();
    if (tid < 192) {
        float* row = &ls[tid * 12];
        float mx = row[0];
        for (int e = 1; e < 12; ++e) mx = fmaxf(mx, row[e]);
        float tmp[12]; float sum = 0.f;
        for (int e = 0; e < 12; ++e) { tmp[e] = expf(row[e] - mx); sum += tmp[e]; }
        float inv = 1.f / sum;
        for (int e = 0; e < 12; ++e)
            attn[(size_t)b * 2304 + tid * 12 + e] = tmp[e] * inv;
    }
}

// ---------------------------------- v = x @ v_w^T (bf16 WMMA) -> vt[b][96][N]
__global__ __launch_bounds__(32)
void vproj_wmma_kernel(const unsigned short* __restrict__ xbf,
                       const unsigned short* __restrict__ wv,
                       float* __restrict__ vt) {
    __shared__ float tile[2][16][17];
    int lane = threadIdx.x;
    int m0 = blockIdx.x * 16, n0 = blockIdx.y * 32;
    int ln = lane & 15, hi = lane >> 4;
    v8f acc0, acc1;
    for (int r = 0; r < 8; ++r) { acc0[r] = 0.f; acc1[r] = 0.f; }
    const unsigned short* pa  = xbf + (size_t)(m0 + ln) * 192 + hi * 8;
    const unsigned short* pb0 = wv  + (size_t)(n0 + ln) * 192 + hi * 8;
    const unsigned short* pb1 = pb0 + 16 * 192;
#pragma unroll
    for (int kc = 0; kc < 6; ++kc) {
        int ko = kc * 32;
        FragU fa, fb0, fb1;
        fa.u4[0]  = *(const uint4*)(pa  + ko);
        fa.u4[1]  = *(const uint4*)(pa  + ko + 16);
        fb0.u4[0] = *(const uint4*)(pb0 + ko);
        fb0.u4[1] = *(const uint4*)(pb0 + ko + 16);
        fb1.u4[0] = *(const uint4*)(pb1 + ko);
        fb1.u4[1] = *(const uint4*)(pb1 + ko + 16);
        acc0 = __builtin_amdgcn_wmma_f32_16x16x32_bf16(false, fa.v, false, fb0.v,
                                                       (short)0, acc0, false, false);
        acc1 = __builtin_amdgcn_wmma_f32_16x16x32_bf16(false, fa.v, false, fb1.v,
                                                       (short)0, acc1, false, false);
    }
#pragma unroll
    for (int r = 0; r < 8; ++r) {
        tile[0][hi * 8 + r][ln] = acc0[r];
        tile[1][hi * 8 + r][ln] = acc1[r];
    }
    __syncthreads();
    int b  = m0 >> 14;                    // 16384 tokens per batch
    int nl = (m0 & 16383) + hi * 8;
    for (int half = 0; half < 2; ++half) {
        int ncol = n0 + half * 16 + ln;
        float* dst = vt + ((size_t)b * 96 + ncol) * 16384 + nl;
#pragma unroll
        for (int r = 0; r < 8; ++r) dst[r] = tile[half][hi * 8 + r][ln];
    }
}

// ------------------------------------------ CPE: depthwise 3x3 + residual add
__global__ __launch_bounds__(256)
void cpe_kernel(const float* __restrict__ vt, const float* __restrict__ cpe_w,
                const float* __restrict__ cpe_b, float* __restrict__ vtot) {
    int tx = threadIdx.x, ty = threadIdx.y;
    int xq = blockIdx.x * 16 + tx, yq = blockIdx.y * 16 + ty;
    int plane = blockIdx.z;
    int b = plane / 96, r = plane % 96;
    const float* src = vt + ((size_t)b * 96 + r) * 16384;
    float w[9];
#pragma unroll
    for (int i = 0; i < 9; ++i) w[i] = cpe_w[r * 9 + i];
    float acc = cpe_b[r];
#pragma unroll
    for (int ky = 0; ky < 3; ++ky) {
        int yy = yq + ky - 1; if ((unsigned)yy >= 128u) continue;
#pragma unroll
        for (int kx = 0; kx < 3; ++kx) {
            int xx = xq + kx - 1; if ((unsigned)xx >= 128u) continue;
            acc += src[yy * 128 + xx] * w[ky * 3 + kx];
        }
    }
    float center = src[yq * 128 + xq];
    vtot[((size_t)b * 96 + r) * 16384 + yq * 128 + xq] = center + acc;
}

// --- out[b,h,d,n] = sum_e attn * vtot ; stored with torch's scrambled reshape
// flat-in-batch index f = d*131072 + h*16384 + n  ->  row n'=f/192, col c=f%192
__global__ __launch_bounds__(256)
void attnout_kernel(const float* __restrict__ attn, const float* __restrict__ vtot,
                    unsigned short* __restrict__ outp) {
    __shared__ float at[2304];
    int b = blockIdx.y, tid = threadIdx.x;
    for (int i = tid; i < 2304; i += 256) at[i] = attn[(size_t)b * 2304 + i];
    __syncthreads();
    int h = tid >> 5;
    int n = blockIdx.x * 32 + (tid & 31);
    float acc[24];
#pragma unroll
    for (int d = 0; d < 24; ++d) acc[d] = 0.f;
    const float* vp = vtot + ((size_t)b * 96 + h * 12) * 16384 + n;
#pragma unroll
    for (int e = 0; e < 12; ++e) {
        float v = vp[(size_t)e * 16384];                // coalesced across lanes
        const float* ar = &at[h * 288 + e];             // at[(h*24+d)*12+e]
#pragma unroll
        for (int d = 0; d < 24; ++d) acc[d] += ar[d * 12] * v;
    }
    unsigned short* ob = outp + (size_t)b * 3145728 + (size_t)h * 16384 + n;
#pragma unroll
    for (int d = 0; d < 24; ++d) ob[(size_t)d * 131072] = f2bf(acc[d]);
}

// --------------------------- final projection (bf16 WMMA) + bias -> f32 d_out
__global__ __launch_bounds__(32)
void proj_wmma_kernel(const unsigned short* __restrict__ opre,
                      const unsigned short* __restrict__ wp,
                      const float* __restrict__ bias, float* __restrict__ out) {
    int lane = threadIdx.x;
    int m0 = blockIdx.x * 16, n0 = blockIdx.y * 32;
    int ln = lane & 15, hi = lane >> 4;
    v8f acc0, acc1;
    for (int r = 0; r < 8; ++r) { acc0[r] = 0.f; acc1[r] = 0.f; }
    const unsigned short* pa  = opre + (size_t)(m0 + ln) * 192 + hi * 8;
    const unsigned short* pb0 = wp   + (size_t)(n0 + ln) * 192 + hi * 8;
    const unsigned short* pb1 = pb0 + 16 * 192;
#pragma unroll
    for (int kc = 0; kc < 6; ++kc) {
        int ko = kc * 32;
        FragU fa, fb0, fb1;
        fa.u4[0]  = *(const uint4*)(pa  + ko);
        fa.u4[1]  = *(const uint4*)(pa  + ko + 16);
        fb0.u4[0] = *(const uint4*)(pb0 + ko);
        fb0.u4[1] = *(const uint4*)(pb0 + ko + 16);
        fb1.u4[0] = *(const uint4*)(pb1 + ko);
        fb1.u4[1] = *(const uint4*)(pb1 + ko + 16);
        acc0 = __builtin_amdgcn_wmma_f32_16x16x32_bf16(false, fa.v, false, fb0.v,
                                                       (short)0, acc0, false, false);
        acc1 = __builtin_amdgcn_wmma_f32_16x16x32_bf16(false, fa.v, false, fb1.v,
                                                       (short)0, acc1, false, false);
    }
    float b0 = bias[n0 + ln], b1 = bias[n0 + 16 + ln];
#pragma unroll
    for (int r = 0; r < 8; ++r) {
        int row = m0 + hi * 8 + r;
        out[(size_t)row * 192 + n0 + ln]      = acc0[r] + b0;   // 64B coalesced
        out[(size_t)row * 192 + n0 + 16 + ln] = acc1[r] + b1;
    }
}

extern "C" void kernel_launch(void* const* d_in, const int* in_sizes, int n_in,
                              void* d_out, int out_size, void* d_ws, size_t ws_size,
                              hipStream_t stream) {
    (void)in_sizes; (void)n_in; (void)out_size; (void)ws_size;
    const float* x      = (const float*)d_in[0];
    const float* red_w  = (const float*)d_in[3];
    const float* red_b  = (const float*)d_in[4];
    const float* dw_w   = (const float*)d_in[5];
    const float* dw_b   = (const float*)d_in[6];
    const float* conv_w = (const float*)d_in[7];
    const float* conv_b = (const float*)d_in[8];
    const float* ln_g   = (const float*)d_in[9];
    const float* ln_b   = (const float*)d_in[10];
    const float* q_w    = (const float*)d_in[11];
    const float* k_w    = (const float*)d_in[12];
    const float* v_w    = (const float*)d_in[13];
    const float* cpe_w  = (const float*)d_in[14];
    const float* cpe_b  = (const float*)d_in[15];
    const float* proj_w = (const float*)d_in[16];
    const float* proj_b = (const float*)d_in[17];

    char* ws = (char*)d_ws;
    unsigned short* xbf  = (unsigned short*)(ws + 0);          // 100663296 B; reused as out_pre
    float* vt            = (float*)(ws + 100663296);           // [16][96][16384]
    float* vtot          = (float*)(ws + 201326592);           // [16][96][16384]
    float* s2            = (float*)(ws + 301989888);           // [16][192][64]
    float* xg            = (float*)(ws + 302776320);           // [16][64][192]
    float* attn          = (float*)(ws + 303562752);           // [16][192][12]
    unsigned short* wvbf = (unsigned short*)(ws + 303710208);  // v_w bf16
    unsigned short* wpbf = (unsigned short*)(ws + 303747072);  // proj_w bf16

    cvt_bf16_kernel<<<65536, 256, 0, stream>>>(x, xbf, 50331648LL);
    cvt_bf16_kernel<<<72, 256, 0, stream>>>(v_w, wvbf, 18432LL);
    cvt_bf16_kernel<<<144, 256, 0, stream>>>(proj_w, wpbf, 36864LL);

    reduce_kernel<<<dim3(8, 8, 16), 192, 0, stream>>>(x, red_w, red_b, s2);
    mix_kernel<<<16, 256, 0, stream>>>(s2, dw_w, dw_b, conv_w, conv_b, ln_g, ln_b, xg);
    qk_attn_kernel<<<16, 256, 0, stream>>>(xg, q_w, k_w, attn);

    vproj_wmma_kernel<<<dim3(16384, 3), 32, 0, stream>>>(xbf, wvbf, vt);
    cpe_kernel<<<dim3(8, 8, 1536), dim3(16, 16), 0, stream>>>(vt, cpe_w, cpe_b, vtot);
    attnout_kernel<<<dim3(512, 16), 256, 0, stream>>>(attn, vtot, xbf /* out_pre bf16 */);
    proj_wmma_kernel<<<dim3(16384, 6), 32, 0, stream>>>(xbf, wpbf, proj_b, (float*)d_out);
}
